// ResNet_36043365548504
// MI455X (gfx1250) — matrix operand order repaired and checked by reading, not compiled, not run on hardware
//
#include <hip/hip_runtime.h>

typedef _Float16 f16;
typedef __attribute__((ext_vector_type(16))) _Float16 v16h;
typedef __attribute__((ext_vector_type(8)))  float    v8f;
typedef __attribute__((ext_vector_type(4)))  unsigned int v4u;
typedef __attribute__((ext_vector_type(4)))  unsigned int u32x4;
typedef __attribute__((ext_vector_type(8)))  unsigned int u32x8;

#define NVOX 300000
#define CCH  32
#define KOFF 27
#define WELEMS (KOFF * CCH * CCH)      // f16 elements per weight layer = 27648
#define WBYTES (WELEMS * 2)            // 55296 bytes
#define WUNITS (WBYTES / 8)            // 6912 8-byte units for the TDM tile
#define SUPERTILES (NVOX / 32)         // 9375 (one wave = 32 voxels)

// ---------------------------------------------------------------------------
// prep: f32 -> f16 activations (with zero pad row at index N), f32 -> f16
// transposed weights wT[k][cout][cin], zero pad row of intermediate buffer.
// ---------------------------------------------------------------------------
__global__ __launch_bounds__(256) void prep_kernel(
    const float* __restrict__ x,
    const float* __restrict__ w0,
    const float* __restrict__ w1,
    f16* __restrict__ xh,     // [(N+1)*32]
    f16* __restrict__ yh,     // [(N+1)*32] (only pad row touched here)
    f16* __restrict__ w0T,    // [27*32*32]
    f16* __restrict__ w1T)    // [27*32*32]
{
  const int gid    = blockIdx.x * blockDim.x + threadIdx.x;
  const int stride = gridDim.x * blockDim.x;

  const int XT = NVOX * CCH;
  for (int i = gid; i < XT; i += stride) xh[i] = (f16)x[i];

  for (int i = gid; i < CCH; i += stride) {
    xh[XT + i] = (f16)0.0f;
    yh[XT + i] = (f16)0.0f;
  }

  for (int t = gid; t < WELEMS; t += stride) {
    int k    = t / (CCH * CCH);
    int r    = t % (CCH * CCH);
    int cout = r / CCH;
    int cin  = r % CCH;
    w0T[t] = (f16)w0[k * CCH * CCH + cin * CCH + cout];
    w1T[t] = (f16)w1[k * CCH * CCH + cin * CCH + cout];
  }
}

// Load a 16x32 f16 A fragment row slice for this lane (contiguous k order, same as B):
// lanes 0-15 carry K=0..15, lanes 16-31 carry K=16..31.
// byteOff is a 32-bit offset from the uniform base -> SADDR+VOFFSET form.
static __device__ __forceinline__ v16h load_a_frag(const char* base,
                                                   unsigned byteOff) {
  v16h a;
  v4u* d = (v4u*)&a;
  d[0] = *(const v4u*)(base + byteOff);
  d[1] = *(const v4u*)(base + byteOff + 16);
  return a;
}

// Load a 32x16 f16 B fragment from LDS weights wT[k][cout][cin]:
// lane l -> column n, lanes 0-15 carry K(cin)=0..15, lanes 16-31 K=16..31.
static __device__ __forceinline__ v16h load_b_frag(const f16* wrow) {
  v16h b;
  v4u* d = (v4u*)&b;
  d[0] = *(const v4u*)(wrow);
  d[1] = *(const v4u*)(wrow + 8);
  return b;
}

// ---------------------------------------------------------------------------
// Sparse conv layer. One wave = 32 voxels (two 16-row A fragments) x 32 output
// channels (two B fragments) -> 4 WMMA per kernel offset, B reused across the
// two voxel tiles. Weights are staged into LDS once per block by the Tensor
// Data Mover (async DMA, TENSORcnt), then read back with ds_load_b128.
// ---------------------------------------------------------------------------
template <bool FIRST>
__global__ __launch_bounds__(256) void spconv_kernel(
    const f16*   __restrict__ src,    // gather source [(N+1)*32] f16
    const int*   __restrict__ nbr,    // [K*N]
    const f16*   __restrict__ wT,     // [27*32*32] transposed f16 weights
    const float* __restrict__ bias,   // [32]
    const float* __restrict__ xres,   // residual (layer 2) or nullptr
    f16*         __restrict__ ydst,   // f16 out (layer 1) or nullptr
    float*       __restrict__ outdst) // f32 out (layer 2) or nullptr
{
  __shared__ __align__(16) f16 wlds[WELEMS];

  const int lane = threadIdx.x & 31;
  const int wave = threadIdx.x >> 5;

  // --- TDM: wave 0 issues one tensor_load_to_lds for the whole 55KB blob ---
  if (wave == 0) {
    const unsigned long long ga = (unsigned long long)(uintptr_t)wT;
    const unsigned int       la = (unsigned int)(uintptr_t)wlds;
    u32x4 g0;
    g0.x = 1u;                                   // count=1 (valid user D#)
    g0.y = la;                                   // lds_addr
    g0.z = (unsigned int)ga;                     // global_addr[31:0]
    g0.w = (unsigned int)(ga >> 32) | (2u << 30);// global_addr[56:32] | type=2
    u32x8 g1;
    g1[0] = (3u << 16);                          // data_size=3 (8B), no extras
    g1[1] = ((unsigned int)(WUNITS & 0xFFFF)) << 16;   // tensor_dim0[15:0]
    g1[2] = (1u << 16);                          // tensor_dim0 hi=0, tensor_dim1=1
    g1[3] = ((unsigned int)WUNITS) << 16;        // tile_dim0 = 6912 (1-D tile)
    g1[4] = 0u;                                  // tile_dim1 = tile_dim2 = 0
    g1[5] = (unsigned int)WUNITS;                // tensor_dim0_stride
    g1[6] = ((unsigned int)(WUNITS & 0xFFFF)) << 16;   // tensor_dim1_stride lo
    g1[7] = 0u;
    asm volatile("tensor_load_to_lds %0, %1" : : "s"(g0), "s"(g1) : "memory");
    __builtin_amdgcn_s_wait_tensorcnt(0);
  }
  __syncthreads();

  const int stile = blockIdx.x * (blockDim.x >> 5) + wave;
  if (stile >= SUPERTILES) return;

  const int vbase = stile * 32;       // 32 voxels: rows [vbase, vbase+32)
  const int m16   = lane & 15;
  const int hi    = lane >> 4;

  const char* srcb   = (const char*)src;
  const unsigned hOff = (unsigned)(hi * 32);   // lane-half byte offset into row

  v8f acc00 = {}, acc01 = {};         // voxels 0..15  x cout 0..15 / 16..31
  v8f acc10 = {}, acc11 = {};         // voxels 16..31 x cout 0..15 / 16..31

  for (int k = 0; k < KOFF; ++k) {
    const int idx0 = nbr[k * NVOX + vbase + m16];
    const int idx1 = nbr[k * NVOX + vbase + 16 + m16];
    // 32-bit byte offsets (feature table < 19.2MB) -> saddr+voffset loads
    const v16h a0 = load_a_frag(srcb, ((unsigned)idx0 << 6) + hOff);
    const v16h a1 = load_a_frag(srcb, ((unsigned)idx1 << 6) + hOff);

    const f16* wk = wlds + k * CCH * CCH;
    const v16h b0 = load_b_frag(wk + m16 * CCH + hi * 16);          // cout 0..15
    const v16h b1 = load_b_frag(wk + (16 + m16) * CCH + hi * 16);   // cout 16..31

    acc00 = __builtin_amdgcn_wmma_f32_16x16x32_f16(false, a0, false, b0,
                                                   (short)0, acc00, false, false);
    acc01 = __builtin_amdgcn_wmma_f32_16x16x32_f16(false, a0, false, b1,
                                                   (short)0, acc01, false, false);
    acc10 = __builtin_amdgcn_wmma_f32_16x16x32_f16(false, a1, false, b0,
                                                   (short)0, acc10, false, false);
    acc11 = __builtin_amdgcn_wmma_f32_16x16x32_f16(false, a1, false, b1,
                                                   (short)0, acc11, false, false);
  }

  const float bn0 = bias[m16];
  const float bn1 = bias[16 + m16];

  if (FIRST) {
    char* yb = (char*)ydst;
#pragma unroll
    for (int r = 0; r < 8; ++r) {
      const int m = r + 8 * hi;
      const unsigned o0 = ((unsigned)(vbase + m) << 6)      + (unsigned)(m16 * 2);
      const unsigned o1 = ((unsigned)(vbase + 16 + m) << 6) + (unsigned)(m16 * 2);
      float v00 = acc00[r] + bn0; v00 = v00 > 0.0f ? v00 : 0.0f;
      float v01 = acc01[r] + bn1; v01 = v01 > 0.0f ? v01 : 0.0f;
      float v10 = acc10[r] + bn0; v10 = v10 > 0.0f ? v10 : 0.0f;
      float v11 = acc11[r] + bn1; v11 = v11 > 0.0f ? v11 : 0.0f;
      *(f16*)(yb + o0)      = (f16)v00;
      *(f16*)(yb + o0 + 32) = (f16)v01;
      *(f16*)(yb + o1)      = (f16)v10;
      *(f16*)(yb + o1 + 32) = (f16)v11;
    }
  } else {
    const char* xb = (const char*)xres;
    char*       ob = (char*)outdst;
#pragma unroll
    for (int r = 0; r < 8; ++r) {
      const int m = r + 8 * hi;
      const unsigned o0 = ((unsigned)(vbase + m) << 7)      + (unsigned)(m16 * 4);
      const unsigned o1 = ((unsigned)(vbase + 16 + m) << 7) + (unsigned)(m16 * 4);
      *(float*)(ob + o0)      = acc00[r] + bn0 + *(const float*)(xb + o0);
      *(float*)(ob + o0 + 64) = acc01[r] + bn1 + *(const float*)(xb + o0 + 64);
      *(float*)(ob + o1)      = acc10[r] + bn0 + *(const float*)(xb + o1);
      *(float*)(ob + o1 + 64) = acc11[r] + bn1 + *(const float*)(xb + o1 + 64);
    }
  }
}

// ---------------------------------------------------------------------------
extern "C" void kernel_launch(void* const* d_in, const int* in_sizes, int n_in,
                              void* d_out, int out_size, void* d_ws, size_t ws_size,
                              hipStream_t stream) {
  const float* x   = (const float*)d_in[0];
  const int*   nbr = (const int*)d_in[1];
  const float* w0  = (const float*)d_in[2];
  const float* b0  = (const float*)d_in[3];
  const float* w1  = (const float*)d_in[4];
  const float* b1  = (const float*)d_in[5];
  float* out = (float*)d_out;

  const size_t featBytes = (size_t)(NVOX + 1) * CCH * sizeof(f16);   // ~19.2 MB
  const size_t featSlot  = (featBytes + 255) & ~(size_t)255;
  const size_t wSlot     = ((size_t)WBYTES + 255) & ~(size_t)255;

  char* ws = (char*)d_ws;
  f16* xh  = (f16*)(ws);
  f16* yh  = (f16*)(ws + featSlot);
  f16* w0T = (f16*)(ws + 2 * featSlot);
  f16* w1T = (f16*)(ws + 2 * featSlot + wSlot);

  prep_kernel<<<4096, 256, 0, stream>>>(x, w0, w1, xh, yh, w0T, w1T);

  const int blocks = (SUPERTILES + 7) / 8;   // 8 waves (32 voxels each) per block
  spconv_kernel<true><<<blocks, 256, 0, stream>>>(xh, nbr, w0T, b0,
                                                  nullptr, yh, nullptr);
  spconv_kernel<false><<<blocks, 256, 0, stream>>>(yh, nbr, w1T, b1,
                                                   x, nullptr, out);
}
